// GraphMoEAttentionRouter_10101763080593
// MI455X (gfx1250) — matrix operand order repaired and checked by reading, not compiled, not run on hardware
//
#include <hip/hip_runtime.h>
#include <hip/hip_bf16.h>
#include <math.h>
#include <stdint.h>

#define N_NODES 8192
#define N_EDGES 131072
#define HID 256
#define N_EXP 8
#define N_GRAPHS 64

typedef __attribute__((ext_vector_type(16))) __bf16 v16bf;
typedef __attribute__((ext_vector_type(8)))  __bf16 v8bf;
typedef __attribute__((ext_vector_type(8)))  float  v8f;

#define WMMA_BF16(a, b, c) \
  __builtin_amdgcn_wmma_f32_16x16x32_bf16(false, (a), false, (b), (short)0, (c), false, false)

// ---------------------------------------------------------------------------
// Async global -> LDS copy of one 16B chunk (CDNA5 GLOBAL_LOAD_ASYNC_TO_LDS).
// LDS address = low 32 bits of the flat shared pointer (ISA 10.2 aperture rule).
// ---------------------------------------------------------------------------
__device__ inline void async_cp16(void* l, const void* g) {
  unsigned loff = (unsigned)(uintptr_t)l;
  asm volatile("global_load_async_to_lds_b128 %0, %1, off"
               :: "v"(loff), "v"(g) : "memory");
}

// Copy CPT*TPB*16 bytes (contiguous) global -> LDS, CPT chunks per thread.
template <int CPT, int TPB>
__device__ inline void stage_lds(__bf16* l, const __bf16* g, int tid) {
#pragma unroll
  for (int i = 0; i < CPT; ++i) {
    int c = i * TPB + tid;
    async_cp16(l + (size_t)c * 8, g + (size_t)c * 8);
  }
}

// ---------------------------------------------------------------------------
// A-fragment loader: 16x32 bf16 A tile, lane<16 row=l K={0..15},
// lane>=16 same row K={16..31}. Row-major source, row stride 256.
// ---------------------------------------------------------------------------
__device__ inline v16bf load_a_frag(const __bf16* __restrict__ A, int row, int kbase, int hi) {
  const __bf16* p = A + (size_t)row * HID + kbase + (hi ? 16 : 0);
  v8bf lo  = *(const v8bf*)p;
  v8bf hi8 = *(const v8bf*)(p + 8);
  return __builtin_shufflevector(lo, hi8, 0,1,2,3,4,5,6,7,8,9,10,11,12,13,14,15);
}

// ---------------------------------------------------------------------------
// Encoder: h = relu(x @ encW + b)
// ---------------------------------------------------------------------------
__global__ __launch_bounds__(256) void k_encoder(const float* __restrict__ x,
                                                 const float* __restrict__ W,
                                                 const float* __restrict__ b,
                                                 float* __restrict__ h,
                                                 __bf16* __restrict__ hbf) {
  int n = blockIdx.x, c = threadIdx.x;
  float acc = b[c];
#pragma unroll
  for (int i = 0; i < 6; ++i) acc += x[n * 6 + i] * W[i * HID + c];
  acc = fmaxf(acc, 0.f);
  h[(size_t)n * HID + c]   = acc;
  hbf[(size_t)n * HID + c] = (__bf16)acc;
}

// ---------------------------------------------------------------------------
// Graph stats / CSR build
// ---------------------------------------------------------------------------
__global__ __launch_bounds__(256) void k_counts(const int* __restrict__ src,
                                                const int* __restrict__ dst,
                                                const int* __restrict__ batch,
                                                int* __restrict__ deg_i,
                                                float* __restrict__ npg,
                                                float* __restrict__ epg) {
  int i = blockIdx.x * blockDim.x + threadIdx.x;
  if (i < N_EDGES) {
    atomicAdd(&deg_i[dst[i]], 1);
    atomicAdd(&epg[batch[src[i]]], 1.0f);
  }
  if (i < N_NODES) atomicAdd(&npg[batch[i]], 1.0f);
}

__global__ __launch_bounds__(1024) void k_scan(const int* __restrict__ deg_i,
                                               int* __restrict__ row_ptr,
                                               float* __restrict__ deg_f) {
  __shared__ int part[1024];
  int t = threadIdx.x, base = t * 8, s = 0;
  int local[8];
#pragma unroll
  for (int i = 0; i < 8; ++i) { local[i] = s; s += deg_i[base + i]; }
  part[t] = s;
  __syncthreads();
  for (int off = 1; off < 1024; off <<= 1) {
    int v = (t >= off) ? part[t - off] : 0;
    __syncthreads();
    part[t] += v;
    __syncthreads();
  }
  int prefix = (t == 0) ? 0 : part[t - 1];
#pragma unroll
  for (int i = 0; i < 8; ++i) row_ptr[base + i] = prefix + local[i];
  if (t == 1023) row_ptr[N_NODES] = part[1023];
#pragma unroll
  for (int i = 0; i < 8; ++i) deg_f[base + i] = fmaxf((float)deg_i[base + i], 1.0f);
}

__global__ __launch_bounds__(256) void k_copy_i(const int* __restrict__ a, int* __restrict__ b, int n) {
  int i = blockIdx.x * blockDim.x + threadIdx.x;
  if (i < n) b[i] = a[i];
}

__global__ __launch_bounds__(256) void k_fill(const int* __restrict__ src,
                                              const int* __restrict__ dst,
                                              int* __restrict__ cursor,
                                              int* __restrict__ col_idx) {
  int i = blockIdx.x * blockDim.x + threadIdx.x;
  if (i < N_EDGES) {
    int p = atomicAdd(&cursor[dst[i]], 1);
    col_idx[p] = src[i];
  }
}

__global__ __launch_bounds__(256) void k_sizefeat(const int* __restrict__ batch,
                                                  const float* __restrict__ npg,
                                                  const float* __restrict__ epg,
                                                  float* __restrict__ sf) {
  int n = blockIdx.x * blockDim.x + threadIdx.x;
  if (n < N_NODES) {
    int g = batch[n];
    sf[2 * n]     = log1pf(npg[g]);
    sf[2 * n + 1] = log1pf(epg[g]);
  }
}

// ---------------------------------------------------------------------------
// Pack ALL 27 [256,256] f32 weight matrices into bf16 WMMA B-fragment order
// in one launch. Matrix m: 0=qW 1=kW 2=vW, 3..18=expWh, 19..26=expWo.
// B 32x16 layout: lane<16 -> N=lane, K=0..15 ; lane>=16 -> N=lane-16, K=16..31.
// ---------------------------------------------------------------------------
__global__ __launch_bounds__(32) void k_pack_all(const float* __restrict__ qW,
                                                 const float* __restrict__ kW,
                                                 const float* __restrict__ vW,
                                                 const float* __restrict__ expWh,
                                                 const float* __restrict__ expWo,
                                                 __bf16* __restrict__ wqPK,
                                                 __bf16* __restrict__ wkPK,
                                                 __bf16* __restrict__ wvPK,
                                                 __bf16* __restrict__ whPK,
                                                 __bf16* __restrict__ woPK) {
  int m    = blockIdx.x >> 7;         // matrix id 0..26
  int tile = blockIdx.x & 127;        // kt*16 + nt
  const float* W;
  __bf16* out;
  if (m == 0)      { W = qW;  out = wqPK; }
  else if (m == 1) { W = kW;  out = wkPK; }
  else if (m == 2) { W = vW;  out = wvPK; }
  else if (m < 19) { W = expWh + (size_t)(m - 3) * 65536;  out = whPK + (size_t)(m - 3) * 65536; }
  else             { W = expWo + (size_t)(m - 19) * 65536; out = woPK + (size_t)(m - 19) * 65536; }

  int kt = tile >> 4, nt = tile & 15;
  int lane = threadIdx.x;
  int ncol  = nt * 16 + (lane & 15);
  int kbase = kt * 32 + ((lane >= 16) ? 16 : 0);
  v16bf t;
#pragma unroll
  for (int e = 0; e < 16; ++e) t[e] = (__bf16)W[(size_t)(kbase + e) * HID + ncol];
  *(v16bf*)(out + ((size_t)tile * 32 + lane) * 16) = t;
}

__global__ __launch_bounds__(32) void k_pack_vt(const __bf16* __restrict__ v, __bf16* __restrict__ out) {
  int tile = blockIdx.x;              // jt*16 + nt, jt in [0,512)
  int jt = tile >> 4, nt = tile & 15;
  int lane = threadIdx.x;
  int ncol  = nt * 16 + (lane & 15);
  int jbase = jt * 32 + ((lane >= 16) ? 16 : 0);
  v16bf t;
#pragma unroll
  for (int e = 0; e < 16; ++e) t[e] = v[(size_t)(jbase + e) * HID + ncol];
  *(v16bf*)(out + ((size_t)tile * 32 + lane) * 16) = t;
}

// ---------------------------------------------------------------------------
// [8192,256] x [256,256] bf16 WMMA GEMM, 64x256 tile per block.
// A-fragments preloaded to registers; B kt-slices async-staged to LDS,
// double-buffered. Main loop touches only LDS + WMMA.
// MODE 0: QKV projection   MODE 1: expert hidden (relu)   MODE 2: weighted combine
// ---------------------------------------------------------------------------
template <int MODE>
__global__ __launch_bounds__(256) void k_gemm(const __bf16* __restrict__ A,
                                              const __bf16* __restrict__ Bpk,
                                              const float* __restrict__ bias,
                                              const float* __restrict__ extraW,
                                              const float* __restrict__ sf,
                                              const float* __restrict__ sparse,
                                              int expert, int accumulate,
                                              float* __restrict__ outF,
                                              __bf16* __restrict__ outBF) {
  __shared__ __bf16 lds_b[2][8192];   // one kt-slice = 16 ntiles * 32 lanes * 16 = 16KB
  int tid = threadIdx.x;
  int lane = tid & 31, wave = tid >> 5;
  int rowGroup = wave >> 1, colHalf = wave & 1;
  int waveM = blockIdx.x * 64 + rowGroup * 16;
  int waveN = colHalf * 128;
  int l15 = lane & 15, hi = lane >> 4;
  int arow = waveM + l15;

  stage_lds<4, 256>(lds_b[0], Bpk, tid);                 // 4 async insts / wave

  // Preload all A-fragments (8 x 8 VGPRs) while B slice 0 streams in.
  v16bf afr[8];
#pragma unroll
  for (int kt = 0; kt < 8; ++kt) afr[kt] = load_a_frag(A, arow, kt * 32, hi);

  v8f acc[8];
#pragma unroll
  for (int i = 0; i < 8; ++i) acc[i] = (v8f){};

  for (int kt = 0; kt < 8; ++kt) {
    int cur = kt & 1;
    if (kt < 7) {
      stage_lds<4, 256>(lds_b[cur ^ 1], Bpk + (size_t)(kt + 1) * 8192, tid);
      asm volatile("s_wait_asynccnt 0x4" ::: "memory");  // current slice done
    } else {
      asm volatile("s_wait_asynccnt 0x0" ::: "memory");
    }
    __syncthreads();
#pragma unroll
    for (int nt = 0; nt < 8; ++nt) {
      int ntile = (waveN >> 4) + nt;
      const v16bf* bp = (const v16bf*)(lds_b[cur] + ((size_t)ntile * 32 + lane) * 16);
      acc[nt] = WMMA_BF16(afr[kt], *bp, acc[nt]);
    }
    __syncthreads();
  }

#pragma unroll
  for (int nt = 0; nt < 8; ++nt) {
    int col = waveN + nt * 16 + l15;
#pragma unroll
    for (int r = 0; r < 8; ++r) {
      int orow = waveM + r + 8 * hi;
      float v = acc[nt][r];
      if (MODE == 0) {
        v += sf[2 * orow] * extraW[256 * HID + col] + sf[2 * orow + 1] * extraW[257 * HID + col] + bias[col];
        outBF[(size_t)orow * HID + col] = (__bf16)v;
      } else if (MODE == 1) {
        v = fmaxf(v + bias[col], 0.f);
        outF[(size_t)orow * HID + col] = v;
      } else {
        v += bias[col];
        float w = sparse[orow * N_EXP + expert];
        float cur2 = accumulate ? outF[(size_t)orow * HID + col] : 0.f;
        outF[(size_t)orow * HID + col] = cur2 + w * v;
      }
    }
  }
}

// ---------------------------------------------------------------------------
// Flash attention: 4 waves/block, 16 Q rows per wave, online softmax.
// K-tile (32 rows x 256d = 16KB contiguous) async-staged to LDS, double buffered,
// shared by all 4 waves. Q fragments live in registers. V fragments pre-packed.
// ---------------------------------------------------------------------------
#define FA_WAVES 4
__global__ __launch_bounds__(FA_WAVES * 32) void k_flash(const __bf16* __restrict__ qbf,
                                                         const __bf16* __restrict__ kbf,
                                                         const __bf16* __restrict__ vtpk,
                                                         float* __restrict__ fused) {
  __shared__ __bf16 lds_k[2][32 * HID];          // 2 x 16KB
  __shared__ __bf16 lds_p[FA_WAVES * 16 * 32];   // P staging, 4KB
  int tid = threadIdx.x;
  int lane = tid & 31, wave = tid >> 5;
  int l15 = lane & 15, hi = lane >> 4;
  int qrow0 = blockIdx.x * (FA_WAVES * 16) + wave * 16;
  __bf16* psh = lds_p + wave * 16 * 32;

  stage_lds<8, FA_WAVES * 32>(lds_k[0], kbf, tid);       // 8 async insts / wave

  // Q A-fragments in registers (reused 16x per j-step)
  v16bf qa[8];
#pragma unroll
  for (int kt = 0; kt < 8; ++kt) qa[kt] = load_a_frag(qbf, qrow0 + l15, kt * 32, hi);

  float m[8], lsum[8], corr[8];
#pragma unroll
  for (int r = 0; r < 8; ++r) { m[r] = -1e30f; lsum[r] = 0.f; }
  v8f oacc[16];
#pragma unroll
  for (int i = 0; i < 16; ++i) oacc[i] = (v8f){};

  const float scale = 0.06225728063f;  // 1/sqrt(258)

  for (int jb = 0; jb < N_NODES; jb += 32) {
    int cur = (jb >> 5) & 1;
    if (jb + 32 < N_NODES) {
      stage_lds<8, FA_WAVES * 32>(lds_k[cur ^ 1], kbf + (size_t)(jb + 32) * HID, tid);
      asm volatile("s_wait_asynccnt 0x8" ::: "memory");  // current K tile done
    } else {
      asm volatile("s_wait_asynccnt 0x0" ::: "memory");
    }
    __syncthreads();

    const __bf16* kbuf = lds_k[cur];
    v8f s0 = (v8f){}, s1 = (v8f){};
#pragma unroll
    for (int kt = 0; kt < 8; ++kt) {
      // K^T B-fragment from LDS: lane<16 -> j=l15, d=kt*32+e ; lane>=16 -> d+=16
      v16bf b0 = *(const v16bf*)(kbuf + (size_t)l15 * HID + kt * 32 + hi * 16);
      v16bf b1 = *(const v16bf*)(kbuf + (size_t)(16 + l15) * HID + kt * 32 + hi * 16);
      s0 = WMMA_BF16(qa[kt], b0, s0);
      s1 = WMMA_BF16(qa[kt], b1, s1);
    }

    // online softmax update (rows r+8*hi; columns across the 16-lane half)
#pragma unroll
    for (int r = 0; r < 8; ++r) {
      float v0 = s0[r] * scale, v1 = s1[r] * scale;
      float tmax = fmaxf(v0, v1);
#pragma unroll
      for (int off = 1; off < 16; off <<= 1) tmax = fmaxf(tmax, __shfl_xor(tmax, off, 32));
      float mn = fmaxf(m[r], tmax);
      float c  = __expf(m[r] - mn);
      float p0 = __expf(v0 - mn), p1 = __expf(v1 - mn);
      float rs = p0 + p1;
#pragma unroll
      for (int off = 1; off < 16; off <<= 1) rs += __shfl_xor(rs, off, 32);
      lsum[r] = lsum[r] * c + rs;
      m[r] = mn;
      corr[r] = c;
      int prow = r + 8 * hi;
      psh[prow * 32 + l15]      = (__bf16)p0;
      psh[prow * 32 + 16 + l15] = (__bf16)p1;
    }
#pragma unroll
    for (int nt = 0; nt < 16; ++nt)
#pragma unroll
      for (int r = 0; r < 8; ++r) oacc[nt][r] *= corr[r];

    asm volatile("s_wait_dscnt 0x0" ::: "memory");

    // Re-read P as A-fragment: lane row=l15, cols {hi*16..+15}
    const __bf16* pp = psh + l15 * 32 + hi * 16;
    v8bf plo = *(const v8bf*)pp;
    v8bf phi = *(const v8bf*)(pp + 8);
    v16bf pa = __builtin_shufflevector(plo, phi, 0,1,2,3,4,5,6,7,8,9,10,11,12,13,14,15);

    int jt = jb >> 5;
#pragma unroll
    for (int nt = 0; nt < 16; ++nt) {
      const v16bf* vb = (const v16bf*)(vtpk + (((size_t)jt * 16 + nt) * 32 + lane) * 16);
      oacc[nt] = WMMA_BF16(pa, *vb, oacc[nt]);
    }
    __syncthreads();
  }

#pragma unroll
  for (int nt = 0; nt < 16; ++nt) {
    int col = nt * 16 + l15;
#pragma unroll
    for (int r = 0; r < 8; ++r) {
      int orow = qrow0 + r + 8 * hi;
      fused[(size_t)orow * HID + col] = oacc[nt][r] / lsum[r];
    }
  }
}

// ---------------------------------------------------------------------------
// Router: logits = fused @ outW + b; renormalized top-2 of softmax
// ---------------------------------------------------------------------------
__global__ __launch_bounds__(256) void k_router(const float* __restrict__ fused,
                                                const float* __restrict__ outW,
                                                const float* __restrict__ outB,
                                                float* __restrict__ sparse) {
  int n = blockIdx.x * blockDim.x + threadIdx.x;
  if (n >= N_NODES) return;
  float lg[N_EXP];
#pragma unroll
  for (int e = 0; e < N_EXP; ++e) lg[e] = outB[e];
  for (int k = 0; k < HID; ++k) {
    float f = fused[(size_t)n * HID + k];
#pragma unroll
    for (int e = 0; e < N_EXP; ++e) lg[e] += f * outW[k * N_EXP + e];
  }
  int i1 = 0;
#pragma unroll
  for (int e = 1; e < N_EXP; ++e) if (lg[e] > lg[i1]) i1 = e;
  int i2 = (i1 == 0) ? 1 : 0;
#pragma unroll
  for (int e = 0; e < N_EXP; ++e) if (e != i1 && lg[e] > lg[i2]) i2 = e;
  float e2 = __expf(lg[i2] - lg[i1]);
  float w1 = 1.f / (1.f + e2), w2 = e2 / (1.f + e2);
#pragma unroll
  for (int e = 0; e < N_EXP; ++e) sparse[n * N_EXP + e] = 0.f;
  sparse[n * N_EXP + i1] = w1;
  sparse[n * N_EXP + i2] = w2;
}

// ---------------------------------------------------------------------------
// Aggregate: A_bf[n,:] = bf16( He[n,:] + (1/deg[n]) * sum_{s in nbrs(n)} He[s,:] )
// ---------------------------------------------------------------------------
__global__ __launch_bounds__(256) void k_aggregate(const float* __restrict__ He,
                                                   const int* __restrict__ row_ptr,
                                                   const int* __restrict__ col_idx,
                                                   const float* __restrict__ deg,
                                                   __bf16* __restrict__ Abf) {
  int n = blockIdx.x, t = threadIdx.x;
  int s0 = row_ptr[n], s1 = row_ptr[n + 1];
  float sum = 0.f;
  for (int e = s0; e < s1; ++e) sum += He[(size_t)col_idx[e] * HID + t];
  float v = He[(size_t)n * HID + t] + sum / deg[n];
  Abf[(size_t)n * HID + t] = (__bf16)v;
}

// ---------------------------------------------------------------------------
extern "C" void kernel_launch(void* const* d_in, const int* in_sizes, int n_in,
                              void* d_out, int out_size, void* d_ws, size_t ws_size,
                              hipStream_t stream) {
  const float* x      = (const float*)d_in[0];
  const int*   eidx   = (const int*)d_in[1];
  const int*   batch  = (const int*)d_in[2];
  const float* encW   = (const float*)d_in[3];
  const float* encB   = (const float*)d_in[4];
  const float* qW     = (const float*)d_in[5];
  const float* qB     = (const float*)d_in[6];
  const float* kW     = (const float*)d_in[7];
  const float* kB     = (const float*)d_in[8];
  const float* vW     = (const float*)d_in[9];
  const float* vB     = (const float*)d_in[10];
  const float* outW   = (const float*)d_in[11];
  const float* outB   = (const float*)d_in[12];
  const float* expWh  = (const float*)d_in[13];
  const float* expBh  = (const float*)d_in[14];
  const float* expWo  = (const float*)d_in[15];
  const float* expBo  = (const float*)d_in[16];
  float* out = (float*)d_out;

  const int* src = eidx;
  const int* dst = eidx + N_EDGES;

  char* ws = (char*)d_ws;
  auto alloc = [&](size_t bytes) -> void* {
    void* p = (void*)ws;
    ws += (bytes + 255) & ~(size_t)255;
    return p;
  };
  const size_t NF = (size_t)N_NODES * HID;
  float*  hF    = (float*)alloc(NF * 4);
  __bf16* hBF   = (__bf16*)alloc(NF * 2);
  __bf16* a0BF  = (__bf16*)alloc(NF * 2);
  __bf16* aBF   = (__bf16*)alloc(NF * 2);
  __bf16* qBF   = (__bf16*)alloc(NF * 2);
  __bf16* kBF   = (__bf16*)alloc(NF * 2);
  __bf16* vBF   = (__bf16*)alloc(NF * 2);
  __bf16* vtPK  = (__bf16*)alloc(NF * 2);
  float*  fused = (float*)alloc(NF * 4);
  float*  heA   = (float*)alloc(NF * 4);
  float*  heB   = (float*)alloc(NF * 4);
  float*  sparse= (float*)alloc((size_t)N_NODES * N_EXP * 4);
  __bf16* wqPK  = (__bf16*)alloc(65536 * 2);
  __bf16* wkPK  = (__bf16*)alloc(65536 * 2);
  __bf16* wvPK  = (__bf16*)alloc(65536 * 2);
  __bf16* whPK  = (__bf16*)alloc((size_t)16 * 65536 * 2);
  __bf16* woPK  = (__bf16*)alloc((size_t)8 * 65536 * 2);
  float*  sf    = (float*)alloc((size_t)N_NODES * 2 * 4);
  float*  npg   = (float*)alloc(N_GRAPHS * 4);
  float*  epg   = (float*)alloc(N_GRAPHS * 4);
  float*  degF  = (float*)alloc(N_NODES * 4);
  int*    degI  = (int*)alloc(N_NODES * 4);
  int*    rowPtr= (int*)alloc((N_NODES + 1) * 4);
  int*    cursor= (int*)alloc(N_NODES * 4);
  int*    colIdx= (int*)alloc(N_EDGES * 4);

  hipMemsetAsync(npg, 0, N_GRAPHS * 4, stream);
  hipMemsetAsync(epg, 0, N_GRAPHS * 4, stream);
  hipMemsetAsync(degI, 0, N_NODES * 4, stream);

  // encoder + graph structure
  k_encoder<<<N_NODES, 256, 0, stream>>>(x, encW, encB, hF, hBF);
  k_counts<<<(N_EDGES + 255) / 256, 256, 0, stream>>>(src, dst, batch, degI, npg, epg);
  k_scan<<<1, 1024, 0, stream>>>(degI, rowPtr, degF);
  k_copy_i<<<(N_NODES + 255) / 256, 256, 0, stream>>>(rowPtr, cursor, N_NODES);
  k_fill<<<(N_EDGES + 255) / 256, 256, 0, stream>>>(src, dst, cursor, colIdx);
  k_sizefeat<<<(N_NODES + 255) / 256, 256, 0, stream>>>(batch, npg, epg, sf);

  // pack all 27 weight matrices to bf16 B-fragments in one launch
  k_pack_all<<<27 * 128, 32, 0, stream>>>(qW, kW, vW, expWh, expWo,
                                          wqPK, wkPK, wvPK, whPK, woPK);

  // q/k/v projections (WMMA), z = [h, size_feat] handled via rank-2 epilogue
  k_gemm<0><<<128, 256, 0, stream>>>(hBF, wqPK, qB, qW, sf, nullptr, 0, 0, nullptr, qBF);
  k_gemm<0><<<128, 256, 0, stream>>>(hBF, wkPK, kB, kW, sf, nullptr, 0, 0, nullptr, kBF);
  k_gemm<0><<<128, 256, 0, stream>>>(hBF, wvPK, vB, vW, sf, nullptr, 0, 0, nullptr, vBF);
  k_pack_vt<<<8192, 32, 0, stream>>>(vBF, vtPK);

  // flash attention + router
  k_flash<<<N_NODES / (FA_WAVES * 16), FA_WAVES * 32, 0, stream>>>(qBF, kBF, vtPK, fused);
  k_router<<<(N_NODES + 255) / 256, 256, 0, stream>>>(fused, outW, outB, sparse);

  // experts: shared layer-0 aggregation, then per-expert towers
  k_aggregate<<<N_NODES, 256, 0, stream>>>(hF, rowPtr, colIdx, degF, a0BF);
  for (int e = 0; e < N_EXP; ++e) {
    const __bf16* wh0 = whPK + (size_t)(e * 2 + 0) * 65536;
    const __bf16* wh1 = whPK + (size_t)(e * 2 + 1) * 65536;
    const __bf16* wo  = woPK + (size_t)e * 65536;
    const float* bh0 = expBh + (size_t)(e * 2 + 0) * HID;
    const float* bh1 = expBh + (size_t)(e * 2 + 1) * HID;
    const float* bo  = expBo + (size_t)e * HID;
    k_gemm<1><<<128, 256, 0, stream>>>(a0BF, wh0, bh0, nullptr, nullptr, nullptr, 0, 0, heA, nullptr);
    k_aggregate<<<N_NODES, 256, 0, stream>>>(heA, rowPtr, colIdx, degF, aBF);
    k_gemm<1><<<128, 256, 0, stream>>>(aBF, wh1, bh1, nullptr, nullptr, nullptr, 0, 0, heB, nullptr);
    k_aggregate<<<N_NODES, 256, 0, stream>>>(heB, rowPtr, colIdx, degF, aBF);
    k_gemm<2><<<128, 256, 0, stream>>>(aBF, wo, bo, nullptr, nullptr, sparse, e, (e > 0) ? 1 : 0, out, nullptr);
  }
}
